// TacticalGNN_55018531062597
// MI455X (gfx1250) — matrix operand order repaired and checked by reading, not compiled, not run on hardware
//
#include <hip/hip_runtime.h>

// ---------------------------------------------------------------------------
// MI455X / gfx1250 fused GNN forward.
// All heavy GEMMs run on v_wmma_f32_16x16x32_f16 (f16 operands, fp32 accum).
// ~26 GFLOP vs <5 MB of HBM traffic -> purely compute bound; everything is
// organized around feeding the WMMA pipes. The O(N^2) pair head is fused so
// the 536 MB intermediate "h" never exists; its A-fragments are built with
// packed-f16 VALU (v_pk_add/max) straight from LDS.
// ---------------------------------------------------------------------------

typedef _Float16 v16h __attribute__((ext_vector_type(16)));
typedef _Float16 v8h  __attribute__((ext_vector_type(8)));
typedef float    v8f  __attribute__((ext_vector_type(8)));

union AFrag { v16h v; v8h h[2]; };

__device__ __forceinline__ v8f wmma_f16(v16h a, v16h b, v8f c) {
  // (neg_a, A, neg_b, B, c_mod, C, reuse_a, reuse_b)
  return __builtin_amdgcn_wmma_f32_16x16x32_f16(false, a, false, b, (short)0, c,
                                                false, false);
}

__device__ __forceinline__ v8h relu8h(v8h x) {
  v8h r;
#pragma unroll
  for (int e = 0; e < 8; ++e) r[e] = x[e] > (_Float16)0 ? x[e] : (_Float16)0;
  return r;
}

// A fragment (16x32, f16): lane<16 -> M=lane, K = {k0..k0+7, k0+16..k0+23};
// lane>=16 -> M=lane-16, K = {k0+8..k0+15, k0+24..k0+31}.
__device__ __forceinline__ v16h load_frag_a(const _Float16* rowBase, int k0, int lane) {
  AFrag f;
  const _Float16* p = rowBase + k0 + ((lane & 16) ? 8 : 0);
  f.h[0] = *(const v8h*)(p);
  f.h[1] = *(const v8h*)(p + 16);
  return f.v;
}

// B fragment (32x16, f16) from row-major W[N,K]: B[k][n] = W[n0+n][k].
// lane<16 -> N=lane, K = {k0..k0+7, k0+16..k0+23}; lane>=16 -> N=lane-16, K = {k0+8..k0+15, k0+24..k0+31}.
__device__ __forceinline__ v16h load_frag_b(const _Float16* W, int ldw, int n0,
                                            int k0, int lane) {
  AFrag f;
  const _Float16* p = W + (size_t)(n0 + (lane & 15)) * ldw + k0 + ((lane & 16) ? 8 : 0);
  f.h[0] = *(const v8h*)(p);
  f.h[1] = *(const v8h*)(p + 16);
  return f.v;
}

// ---------------------------------------------------------------------------
// Utility kernels
// ---------------------------------------------------------------------------
__global__ void k_zero_f32(float* p, int n) {
  for (int i = blockIdx.x * blockDim.x + threadIdx.x; i < n; i += gridDim.x * blockDim.x)
    p[i] = 0.f;
}

__global__ void k_copy_f32(const float* s, float* d, int n) {
  for (int i = blockIdx.x * blockDim.x + threadIdx.x; i < n; i += gridDim.x * blockDim.x)
    d[i] = s[i];
}

// f32 [rows, scols] -> f16 [rows, dcols] zero-padded (dcols multiple of 32)
__global__ void k_cvt_pad(const float* src, _Float16* dst, int rows, int scols, int dcols) {
  int n = rows * dcols;
  for (int i = blockIdx.x * blockDim.x + threadIdx.x; i < n; i += gridDim.x * blockDim.x) {
    int r = i / dcols, c = i - r * dcols;
    dst[i] = (c < scols) ? (_Float16)src[(size_t)r * scols + c] : (_Float16)0.f;
  }
}

// concat(x[rows,xcols] , agg[rows,128]) -> f16 [rows, dcols] zero padded
__global__ void k_concat_f16(const float* x, int xcols, const float* agg,
                             _Float16* dst, int dcols, int rows) {
  int n = rows * dcols;
  for (int i = blockIdx.x * blockDim.x + threadIdx.x; i < n; i += gridDim.x * blockDim.x) {
    int r = i / dcols, c = i - r * dcols;
    float v = 0.f;
    if (c < xcols)            v = x[(size_t)r * xcols + c];
    else if (c < xcols + 128) v = agg[(size_t)r * 128 + (c - xcols)];
    dst[i] = (_Float16)v;
  }
}

// Fold edge-feature branch of m1 through et:  Wce = We @ et.w,  bcomb = m1.b + We @ et.b
__global__ void k_edge_combine(const float* m1w, const float* m1b,
                               const float* etw, const float* etb,
                               float* Wce, float* bcomb) {
  int n = threadIdx.x;  // 128 threads
  float acc[4] = {0.f, 0.f, 0.f, 0.f};
  float bb = 0.f;
  for (int h = 0; h < 128; ++h) {
    float we = m1w[(size_t)n * 256 + 128 + h];
    bb += we * etb[h];
#pragma unroll
    for (int q = 0; q < 4; ++q) acc[q] += we * etw[h * 4 + q];
  }
#pragma unroll
  for (int q = 0; q < 4; ++q) Wce[n * 4 + q] = acc[q];
  bcomb[n] = m1b[n] + bb;
}

// ---------------------------------------------------------------------------
// Generic WMMA GEMM:  out = act(A[M,K]_f16 @ W[N,K]_f16^T + bias)
// Each wave owns 4 M-tiles x 1 N-tile so every B fragment feeds 4 WMMAs
// (B-load amortization; M must be a multiple of 64). K padded to mult of 32.
// ---------------------------------------------------------------------------
__global__ __launch_bounds__(128)
void k_gemm_wmma(const _Float16* __restrict__ A, int lda,
                 const _Float16* __restrict__ W, int ldw,
                 const float* __restrict__ bias,
                 float* outF, _Float16* outH, int ldo,
                 int M, int N, int K, int relu) {
  int wave = (blockIdx.x * blockDim.x + threadIdx.x) >> 5;
  int lane = threadIdx.x & 31;
  int ntiles = N >> 4;
  int total = (M >> 6) * ntiles;
  if (wave >= total) return;
  int mg = wave / ntiles, nt = wave - mg * ntiles;
  int m0 = mg << 6, n0 = nt << 4;

  v8f acc[4] = {{}, {}, {}, {}};
  const _Float16* aRow = A + (size_t)(m0 + (lane & 15)) * lda;
  for (int k0 = 0; k0 < K; k0 += 32) {
    v16h bf = load_frag_b(W, ldw, n0, k0, lane);
#pragma unroll
    for (int t = 0; t < 4; ++t)
      acc[t] = wmma_f16(load_frag_a(aRow + (size_t)(t << 4) * lda, k0, lane), bf, acc[t]);
  }

  int col = n0 + (lane & 15);
  float bv = bias ? bias[col] : 0.f;
#pragma unroll
  for (int t = 0; t < 4; ++t) {
    int rbase = m0 + (t << 4) + ((lane & 16) ? 8 : 0);
#pragma unroll
    for (int r = 0; r < 8; ++r) {
      float v = acc[t][r] + bv;
      if (relu) v = fmaxf(v, 0.f);
      size_t o = (size_t)(rbase + r) * ldo + col;
      if (outF) outF[o] = v;
      if (outH) outH[o] = (_Float16)v;
    }
  }
}

// ---------------------------------------------------------------------------
// Fused edge MLP + scatter:  per wave, 16 edges.
//   premsg = relu( node_emb[src] @ Wn^T + ef @ Wce^T + bcomb )   (WMMA + init)
//   msg    = premsg @ m2^T + m2.b                                 (WMMA)
//   agg[dst] += msg                                               (f32 atomics)
// ---------------------------------------------------------------------------
__global__ __launch_bounds__(128)
void k_edge_mlp(const _Float16* __restrict__ nodeEmbH,
                const int* __restrict__ src, const int* __restrict__ dst,
                const float* __restrict__ ef,
                const float* __restrict__ Wce, const float* __restrict__ bcomb,
                const _Float16* __restrict__ m1wH,   // [128,256] (node half used)
                const _Float16* __restrict__ m2wH,   // [128,128]
                const float* __restrict__ m2b,
                float* agg, int nEdges) {
  __shared__ int   sSrc[4][16], sDst[4][16];
  __shared__ float sEf[4][16][4];
  __shared__ __align__(16) _Float16 sPre[4][16][128];

  int w = threadIdx.x >> 5, lane = threadIdx.x & 31;
  int e0 = (blockIdx.x * 4 + w) * 16;
  if (e0 >= nEdges) return;

  if (lane < 16) { sSrc[w][lane] = src[e0 + lane]; sDst[w][lane] = dst[e0 + lane]; }
  {
    int t = lane * 2;
    sEf[w][t >> 2][t & 3] = ef[(size_t)e0 * 4 + t];
    t++;
    sEf[w][t >> 2][t & 3] = ef[(size_t)e0 * 4 + t];
  }

  int colL = lane & 15, rb = (lane & 16) ? 8 : 0;
  const _Float16* aRow = nodeEmbH + (size_t)sSrc[w][colL] * 128;

  // GEMM 1: node part via WMMA, edge part + bias folded into accumulator init
  for (int nt = 0; nt < 8; ++nt) {
    int n0 = nt << 4, col = n0 + colL;
    v8f acc;
#pragma unroll
    for (int r = 0; r < 8; ++r) {
      float v = bcomb[col];
#pragma unroll
      for (int q = 0; q < 4; ++q) v += sEf[w][rb + r][q] * Wce[col * 4 + q];
      acc[r] = v;
    }
    for (int k0 = 0; k0 < 128; k0 += 32)
      acc = wmma_f16(load_frag_a(aRow, k0, lane), load_frag_b(m1wH, 256, n0, k0, lane), acc);
#pragma unroll
    for (int r = 0; r < 8; ++r)
      sPre[w][rb + r][col] = (_Float16)fmaxf(acc[r], 0.f);
  }

  // GEMM 2: msg = premsg @ m2^T, scatter-add to agg[dst]
  const _Float16* pRow = &sPre[w][colL][0];
  for (int nt = 0; nt < 8; ++nt) {
    int n0 = nt << 4, col = n0 + colL;
    v8f acc = {};
    for (int k0 = 0; k0 < 128; k0 += 32)
      acc = wmma_f16(load_frag_a(pRow, k0, lane), load_frag_b(m2wH, 128, n0, k0, lane), acc);
    float bv = m2b[col];
#pragma unroll
    for (int r = 0; r < 8; ++r)
      atomicAdd(&agg[(size_t)sDst[w][rb + r] * 128 + col], acc[r] + bv);
  }
}

// ---------------------------------------------------------------------------
// LayerNorm + optional residual + ReLU (one wave per node row)
// ---------------------------------------------------------------------------
__global__ __launch_bounds__(128)
void k_ln_res_relu(const float* __restrict__ pre, const float* __restrict__ g,
                   const float* __restrict__ b, float* x, _Float16* xh, int useRes) {
  int wave = (blockIdx.x * blockDim.x + threadIdx.x) >> 5;
  int lane = threadIdx.x & 31;
  if (wave >= 1024) return;
  const float* row = pre + (size_t)wave * 128;
  float v[4], s = 0.f, s2 = 0.f;
#pragma unroll
  for (int t = 0; t < 4; ++t) { v[t] = row[lane + 32 * t]; s += v[t]; s2 += v[t] * v[t]; }
#pragma unroll
  for (int off = 16; off > 0; off >>= 1) {
    s  += __shfl_xor(s,  off, 32);
    s2 += __shfl_xor(s2, off, 32);
  }
  float mu  = s * (1.f / 128.f);
  float var = s2 * (1.f / 128.f) - mu * mu;
  float inv = rsqrtf(var + 1e-5f);
#pragma unroll
  for (int t = 0; t < 4; ++t) {
    int c = lane + 32 * t;
    float o = (v[t] - mu) * inv * g[c] + b[c];
    if (useRes) o += x[(size_t)wave * 128 + c];
    o = fmaxf(o, 0.f);
    x [(size_t)wave * 128 + c] = o;
    xh[(size_t)wave * 128 + c] = (_Float16)o;
  }
}

// ---------------------------------------------------------------------------
// Head finals (tiny)
// ---------------------------------------------------------------------------
__global__ void k_move_final(const float* __restrict__ h2, const float* __restrict__ w,
                             const float* __restrict__ b, float* out) {
  int n = blockIdx.x * blockDim.x + threadIdx.x;
  if (n >= 1024) return;
  const float* hr = h2 + (size_t)n * 64;
  float lg[4];
#pragma unroll
  for (int j = 0; j < 4; ++j) {
    float s = b[j];
    for (int c = 0; c < 64; ++c) s += w[j * 64 + c] * hr[c];
    lg[j] = s;
  }
  float mx = fmaxf(fmaxf(lg[0], lg[1]), fmaxf(lg[2], lg[3]));
  float e[4], se = 0.f;
#pragma unroll
  for (int j = 0; j < 4; ++j) { e[j] = __expf(lg[j] - mx); se += e[j]; }
#pragma unroll
  for (int j = 0; j < 4; ++j) {
    out[(size_t)n * 4 + j] = lg[j];
    float p = e[j] / se;
    out[4096 + (size_t)n * 4 + j] = fminf(fmaxf(p, 0.001f), 0.999f);
  }
}

__global__ void k_shoot_final(const float* __restrict__ h2, const float* __restrict__ w,
                              const float* __restrict__ b, float* out) {
  int n = blockIdx.x * blockDim.x + threadIdx.x;
  if (n >= 1024) return;
  const float* hr = h2 + (size_t)n * 64;
  float s = b[0];
  for (int c = 0; c < 64; ++c) s += w[c] * hr[c];
  out[n] = fminf(fmaxf(s, -10.f), 10.f);
}

// ---------------------------------------------------------------------------
// Fused pairwise target head. Block = 8 waves = 8 i-values sharing one 16-j
// tile. xab (= xa + t0.b, f16) and xb (f16) live in LDS; the WMMA A-fragment
// h = relu(xab[i]+xb[j]) is built with packed-f16 adds/max (v_pk_*_f16), then
// 4 N-tiles x 4 K-steps of WMMA vs t1.w, relu + dot(w3) reduced through a
// per-wave LDS scratch. The 536 MB "h" never materializes.
// ---------------------------------------------------------------------------
__global__ __launch_bounds__(256)
void k_target_pairs(const _Float16* __restrict__ xabH,  // [1024,128] xa + t0.b
                    const _Float16* __restrict__ xbH,   // [1024,128]
                    const _Float16* __restrict__ t1wH,  // [64,128]
                    const float* __restrict__ t1b,
                    const float* __restrict__ t2w, const float* __restrict__ t2b,
                    float* outTg) {
  __shared__ __align__(16) _Float16 sXb[16][128];
  __shared__ __align__(16) _Float16 sXab[8][128];
  __shared__ __align__(16) _Float16 sW[64 * 128];
  __shared__ float sW3[64];
  __shared__ float sT1b[64];
  __shared__ float sH[8][16][64];

  int wv = threadIdx.x >> 5, lane = threadIdx.x & 31;
  int jt = blockIdx.x & 63, ig = blockIdx.x >> 6;
  int j0 = jt << 4, i = ig * 8 + wv;

  for (int t = threadIdx.x; t < 16 * 128; t += 256) sXb[t >> 7][t & 127] = xbH[(size_t)j0 * 128 + t];
  for (int t = threadIdx.x; t < 64; t += 256) { sW3[t] = t2w[t]; sT1b[t] = t1b[t]; }
  for (int t = threadIdx.x; t < 64 * 128; t += 256) sW[t] = t1wH[t];
  for (int t = lane; t < 128; t += 32) sXab[wv][t] = xabH[(size_t)i * 128 + t];
  __syncthreads();

  v8f acc[4] = {{}, {}, {}, {}};
  int m = lane & 15;
  for (int k0 = 0; k0 < 128; k0 += 32) {
    int kb = k0 + ((lane & 16) ? 8 : 0);
    AFrag a;
    a.h[0] = relu8h(*(const v8h*)&sXab[wv][kb]      + *(const v8h*)&sXb[m][kb]);
    a.h[1] = relu8h(*(const v8h*)&sXab[wv][kb + 16] + *(const v8h*)&sXb[m][kb + 16]);
#pragma unroll
    for (int nt = 0; nt < 4; ++nt)
      acc[nt] = wmma_f16(a.v, load_frag_b(sW, 128, nt << 4, k0, lane), acc[nt]);
  }

  int colL = lane & 15, rb = (lane & 16) ? 8 : 0;
#pragma unroll
  for (int nt = 0; nt < 4; ++nt)
#pragma unroll
    for (int r = 0; r < 8; ++r) {
      float v = acc[nt][r] + sT1b[(nt << 4) + colL];
      sH[wv][rb + r][(nt << 4) + colL] = fmaxf(v, 0.f);
    }
  __syncthreads();

  if (lane < 16) {
    int j = j0 + lane;
    float s = t2b[0];
    for (int c = 0; c < 64; ++c) s += sH[wv][lane][c] * sW3[c];
    s = fminf(fmaxf(s, -10.f), 10.f);
    if (j != i) {
      int p = i * 1023 + (j - (j > i ? 1 : 0));
      outTg[p] = s;
    }
  }
}

// ---------------------------------------------------------------------------
// Host side
// ---------------------------------------------------------------------------
extern "C" void kernel_launch(void* const* d_in, const int* in_sizes, int n_in,
                              void* d_out, int out_size, void* d_ws, size_t ws_size,
                              hipStream_t stream) {
  (void)in_sizes; (void)n_in; (void)out_size; (void)ws_size;
  const int NN = 1024, NE = 32768;

  const float* nf  = (const float*)d_in[0];
  const int*   ei  = (const int*)d_in[1];
  const float* efe = (const float*)d_in[2];
  const int* src = ei;
  const int* dst = ei + NE;

  // params leaves (jax tree-flatten: sorted dict keys); base = 3
  // per conv: et.b, et.w, ln_b, ln_g, m1.b, m1.w, m2.b, m2.w, nt.b, nt.w, u1.b, u1.w, u2.b, u2.w
  enum { ETB = 0, ETW, LNB, LNG, M1B, M1W, M2B, M2W, NTB, NTW, U1B, U1W, U2B, U2W };
  auto P = [&](int idx) { return (const float*)d_in[idx]; };
  auto CV = [&](int l, int which) { return P(3 + l * 14 + which); };
  const int HB = 3 + 4 * 14;  // 59: move0.b, move0.w, move1.b, move1.w, move2.b, move2.w,
                              // shoot(6), target(6)

  // workspace carve
  char* wp = (char*)d_ws;
  auto carve = [&](size_t bytes) -> void* {
    void* r = (void*)wp; wp += (bytes + 255) & ~(size_t)255; return r;
  };
  float* xF    = (float*)carve(NN * 128 * 4);
  float* preF  = (float*)carve(NN * 128 * 4);
  float* agg   = (float*)carve(NN * 128 * 4);
  float* h2F   = (float*)carve(NN * 64 * 4);
  float* Wce   = (float*)carve(128 * 4 * 4);
  float* bcomb = (float*)carve(128 * 4);
  _Float16* xH       = (_Float16*)carve(NN * 128 * 2);
  _Float16* x0H      = (_Float16*)carve(NN * 32 * 2);
  _Float16* nodeEmbH = (_Float16*)carve(NN * 128 * 2);
  _Float16* updH     = (_Float16*)carve(NN * 256 * 2);
  _Float16* hidH     = (_Float16*)carve(NN * 128 * 2);
  _Float16* h1H      = (_Float16*)carve(NN * 128 * 2);
  _Float16* xabH     = (_Float16*)carve(NN * 128 * 2);
  _Float16* xbH      = (_Float16*)carve(NN * 128 * 2);
  _Float16* ntwH = (_Float16*)carve(128 * 128 * 2);
  _Float16* m1wH = (_Float16*)carve(128 * 256 * 2);
  _Float16* m2wH = (_Float16*)carve(128 * 128 * 2);
  _Float16* u1wH = (_Float16*)carve(128 * 256 * 2);
  _Float16* u2wH = (_Float16*)carve(128 * 128 * 2);
  _Float16* w0H  = (_Float16*)carve(128 * 128 * 2);
  _Float16* w1H  = (_Float16*)carve(64 * 128 * 2);
  _Float16* t0wH = (_Float16*)carve(128 * 256 * 2);
  _Float16* t1wH = (_Float16*)carve(64 * 128 * 2);

  float* out = (float*)d_out;
  float* out_target = out + 9216;
  float* out_x      = out + 1056768;

  auto cvt = [&](const float* s, _Float16* d, int rows, int sc, int dc) {
    int n = rows * dc;
    k_cvt_pad<<<(n + 255) / 256, 256, 0, stream>>>(s, d, rows, sc, dc);
  };
  auto gemm = [&](const _Float16* A, int lda, const _Float16* W, int ldw,
                  const float* bias, float* oF, _Float16* oH, int ldo,
                  int M, int N, int K, int relu) {
    int waves = (M / 64) * (N / 16);
    int blocks = (waves * 32 + 127) / 128;
    k_gemm_wmma<<<blocks, 128, 0, stream>>>(A, lda, W, ldw, bias, oF, oH, ldo, M, N, K, relu);
  };

  // initial padded f16 node features
  cvt(nf, x0H, NN, 14, 32);

  for (int l = 0; l < 4; ++l) {
    int cur  = l ? 128 : 14;
    int curp = l ? 128 : 32;   // padded K for nt
    int uk   = cur + 128;
    int ukp  = l ? 256 : 160;  // padded K for u1

    cvt(CV(l, NTW), ntwH, 128, cur, curp);
    cvt(CV(l, M1W), m1wH, 128, 256, 256);
    cvt(CV(l, M2W), m2wH, 128, 128, 128);
    cvt(CV(l, U1W), u1wH, 128, uk, ukp);
    cvt(CV(l, U2W), u2wH, 128, 128, 128);
    k_edge_combine<<<1, 128, 0, stream>>>(CV(l, M1W), CV(l, M1B), CV(l, ETW), CV(l, ETB),
                                          Wce, bcomb);

    // node_emb = x @ nt.w^T + nt.b  (f16 out for the edge gather)
    gemm(l ? xH : x0H, curp, ntwH, curp, CV(l, NTB), nullptr, nodeEmbH, 128,
         NN, 128, curp, 0);

    k_zero_f32<<<256, 256, 0, stream>>>(agg, NN * 128);
    k_edge_mlp<<<NE / 64, 128, 0, stream>>>(nodeEmbH, src, dst, efe, Wce, bcomb,
                                            m1wH, m2wH, CV(l, M2B), agg, NE);

    k_concat_f16<<<(NN * ukp + 255) / 256, 256, 0, stream>>>(
        l ? xF : nf, cur, agg, updH, ukp, NN);
    gemm(updH, ukp, u1wH, ukp, CV(l, U1B), nullptr, hidH, 128, NN, 128, ukp, 1);
    gemm(hidH, 128, u2wH, 128, CV(l, U2B), preF, nullptr, 128, NN, 128, 128, 0);
    k_ln_res_relu<<<256, 128, 0, stream>>>(preF, CV(l, LNG), CV(l, LNB), xF, xH, l > 0);
  }

  // ---- move head ----
  cvt(P(HB + 1), w0H, 128, 128, 128);
  gemm(xH, 128, w0H, 128, P(HB + 0), nullptr, h1H, 128, NN, 128, 128, 1);
  cvt(P(HB + 3), w1H, 64, 128, 128);
  gemm(h1H, 128, w1H, 128, P(HB + 2), h2F, nullptr, 64, NN, 64, 128, 1);
  k_move_final<<<4, 256, 0, stream>>>(h2F, P(HB + 5), P(HB + 4), out);

  // ---- shoot head ----
  cvt(P(HB + 7), w0H, 128, 128, 128);
  gemm(xH, 128, w0H, 128, P(HB + 6), nullptr, h1H, 128, NN, 128, 128, 1);
  cvt(P(HB + 9), w1H, 64, 128, 128);
  gemm(h1H, 128, w1H, 128, P(HB + 8), h2F, nullptr, 64, NN, 64, 128, 1);
  k_shoot_final<<<4, 256, 0, stream>>>(h2F, P(HB + 11), P(HB + 10), out + 8192);

  // ---- target head ----
  cvt(P(HB + 13), t0wH, 128, 256, 256);               // t0.w [128,256]
  // xab = x @ wa^T + t0.b (bias folded here), xb = x @ wb^T -- both f16 out
  gemm(xH, 128, t0wH, 256, P(HB + 12), nullptr, xabH, 128, NN, 128, 128, 0);
  gemm(xH, 128, t0wH + 128, 256, nullptr, nullptr, xbH, 128, NN, 128, 128, 0);
  cvt(P(HB + 15), t1wH, 64, 128, 128);                // t1.w [64,128]
  k_target_pairs<<<64 * 128, 256, 0, stream>>>(xabH, xbH, t1wH,
                                               P(HB + 14), P(HB + 17), P(HB + 16),
                                               out_target);

  // ---- final node embeddings ----
  k_copy_f32<<<512, 256, 0, stream>>>(xF, out_x, NN * 128);
}
